// GraphPropagationBlock_79568564125726
// MI455X (gfx1250) — matrix operand order repaired and checked by reading, not compiled, not run on hardware
//
#include <hip/hip_runtime.h>
#include <hip/hip_bf16.h>

// ---------------- shapes ----------------
constexpr int BSZ   = 64;
constexpr int NTOK  = 197;
constexpr int CDIM  = 768;
constexpr int NH    = 12;
constexpr int HD    = 64;
constexpr int MLPH  = 3072;
constexpr int NPROP = 40;
constexpr int NKEPT = NTOK - 1 - NPROP;   // 156
constexpr int NPRUNE= NTOK - NPROP;       // 157
constexpr float ALPHA = 0.1f;

// ---------------- WMMA types ----------------
typedef __attribute__((ext_vector_type(16))) __bf16 v16bf;
typedef __attribute__((ext_vector_type(8)))  __bf16 v8bf;
typedef __attribute__((ext_vector_type(2)))  __bf16 v2bf;
typedef __attribute__((ext_vector_type(8)))  float  v8f;
typedef __attribute__((ext_vector_type(2)))  float  v2f;

__device__ inline __bf16 f2bf(float f) { return (__bf16)f; }

// exact (lo,hi) pairing -> single v_cvt_pk_bf16_f32, no repacking
__device__ inline unsigned pack2bf(float a, float b) {
  v2f f; f.x = a; f.y = b;
  v2bf c = __builtin_convertvector(f, v2bf);
  unsigned u; __builtin_memcpy(&u, &c, 4);
  return u;
}

// =====================================================================
// Batched bf16-WMMA GEMM:  C[z] = A[z](MxK,f32) * B[z](KxN) (+bias)(+resid)
// BT=true : B[k][n] = W[n*ldb + k]     BT=false: B[k][n] = W[k*ldb + n]
// AV4: A rows are float4-aligned (lda % 4 == 0)
// EPI: 0 plain, 2 +bias+resid, 3 gelu(acc+bias)
// Block: 256 threads = 8 waves; tile 128(M) x 64(N), K-step 32.
// =====================================================================
constexpr int BM = 128, BN = 64, BK = 32;

template<int EPI, bool BT, bool AV4>
__global__ __launch_bounds__(256)
void gemm_wmma(const float* __restrict__ A, long long sA, int lda,
               const float* __restrict__ Bm, long long sB, int ldb,
               float* __restrict__ Cm, long long sC, int ldc,
               const float* __restrict__ bias,
               const float* __restrict__ resid, int ldr,
               int M, int Nn, int K)
{
  __shared__ __bf16 Alds[BM][BK];        // 8 KB
  __shared__ __bf16 Blds[BK][BN + 16];   // padded rows (160B)

  const int z = blockIdx.z;
  A  += (long long)z * sA;
  Bm += (long long)z * sB;
  Cm += (long long)z * sC;

  const int m0 = blockIdx.y * BM;
  const int n0 = blockIdx.x * BN;
  const int tid  = threadIdx.x;
  const int lane = tid & 31;
  const int wave = tid >> 5;

  const bool nfull = (n0 + BN <= Nn);

  // staging coordinates
  const int ar = tid >> 1;            // A: row (128 rows, 2 threads each)
  const int ac = (tid & 1) * 16;      // A: col start (16 floats)
  const int bn_t = tid & 63;          // B (BT): n
  const int bk_t = (tid >> 6) * 8;    // B (BT): k start
  const int bk_d = tid >> 3;          // B (!BT): k
  const int bn_d = (tid & 7) * 8;     // B (!BT): n start

  // branch-free row/col clamps for M/N edges (main K loop)
  const int  gmA   = m0 + ar;
  const bool amask = gmA < M;
  const float* Arow = A + (long long)(amask ? gmA : (M - 1)) * lda;

  const int  gnB   = n0 + bn_t;                    // BT only
  const bool bmask = gnB < Nn;
  const float* Brow_t = BT ? (Bm + (long long)(bmask ? gnB : (Nn - 1)) * ldb) : nullptr;

  v8f acc[4];
  #pragma unroll
  for (int i = 0; i < 4; ++i) acc[i] = 0.f;

  const int mloc = wave * 16 + (lane & 15);
  const int half = lane >> 4;

  const int Kmain = K & ~(BK - 1);

  for (int k0 = 0; k0 < Kmain; k0 += BK) {
    // ---------------- A tile: unguarded loads from clamped row ----------------
    {
      const float* ap = Arow + k0 + ac;
      float av[16];
      if (AV4) {
        const float4* apv = (const float4*)ap;
        #pragma unroll
        for (int g = 0; g < 4; ++g) {
          const float4 f = apv[g];
          av[g*4+0] = f.x; av[g*4+1] = f.y; av[g*4+2] = f.z; av[g*4+3] = f.w;
        }
      } else {
        #pragma unroll
        for (int i = 0; i < 16; ++i) av[i] = ap[i];
      }
      uint4 plo, phi;
      plo.x = pack2bf(av[ 0], av[ 1]); plo.y = pack2bf(av[ 2], av[ 3]);
      plo.z = pack2bf(av[ 4], av[ 5]); plo.w = pack2bf(av[ 6], av[ 7]);
      phi.x = pack2bf(av[ 8], av[ 9]); phi.y = pack2bf(av[10], av[11]);
      phi.z = pack2bf(av[12], av[13]); phi.w = pack2bf(av[14], av[15]);
      if (!amask) { plo = uint4{0,0,0,0}; phi = uint4{0,0,0,0}; }
      *(uint4*)&Alds[ar][ac]     = plo;   // 16B-aligned wide LDS stores
      *(uint4*)&Alds[ar][ac + 8] = phi;
      if (k0 + 2*BK <= K) __builtin_prefetch(ap + BK, 0, 0);
    }
    // ---------------- B tile ----------------
    if (BT) {
      const float4* bpv = (const float4*)(Brow_t + k0 + bk_t);
      const float4 f0 = bpv[0], f1 = bpv[1];
      const float bv[8] = {f0.x,f0.y,f0.z,f0.w,f1.x,f1.y,f1.z,f1.w};
      #pragma unroll
      for (int i = 0; i < 8; ++i)
        Blds[bk_t + i][bn_t] = f2bf(bmask ? bv[i] : 0.f);   // transpose scatter
    } else if (nfull) {
      const float4* bpv = (const float4*)(Bm + (long long)(k0 + bk_d) * ldb + (n0 + bn_d));
      const float4 f0 = bpv[0], f1 = bpv[1];
      uint4 p;
      p.x = pack2bf(f0.x, f0.y); p.y = pack2bf(f0.z, f0.w);
      p.z = pack2bf(f1.x, f1.y); p.w = pack2bf(f1.z, f1.w);
      *(uint4*)&Blds[bk_d][bn_d] = p;
    } else {
      #pragma unroll
      for (int i = 0; i < 8; ++i) {
        const int gn  = n0 + bn_d + i;
        const int gnc = gn < Nn ? gn : (Nn - 1);
        const float v = Bm[(long long)(k0 + bk_d) * ldb + gnc];
        Blds[bk_d][bn_d + i] = f2bf(gn < Nn ? v : 0.f);
      }
    }
    __syncthreads();

    // ---- fragments + 4 WMMAs ----
    {
      v8bf alo = *(const v8bf*)&Alds[mloc][half * 8];
      v8bf ahi = *(const v8bf*)&Alds[mloc][16 + half * 8];
      v16bf afrag = __builtin_shufflevector(alo, ahi,
          0,1,2,3,4,5,6,7,8,9,10,11,12,13,14,15);
      #pragma unroll
      for (int nt = 0; nt < 4; ++nt) {
        v16bf bfrag;
        #pragma unroll
        for (int s = 0; s < 16; ++s)
          bfrag[s] = Blds[s < 8 ? half * 8 + s : 16 + half * 8 + (s - 8)][nt * 16 + (lane & 15)];
        acc[nt] = __builtin_amdgcn_wmma_f32_16x16x32_bf16(
            false, afrag, false, bfrag, (short)0, acc[nt], false, false);
      }
    }
    __syncthreads();
  }

  // ---------------- K tail (K % 32 != 0): fully clamped scalar ----------------
  if (Kmain < K) {
    const int k0 = Kmain;
    #pragma unroll
    for (int i = 0; i < 16; ++i) {
      const int gk  = k0 + ac + i;
      const int gkc = gk < K ? gk : (K - 1);
      const float v = Arow[gkc];
      Alds[ar][ac + i] = f2bf((amask && gk < K) ? v : 0.f);
    }
    {
      const int r = tid >> 3;
      const int c = (tid & 7) * 8;
      const int gk  = k0 + r;
      const int gkc = gk < K ? gk : (K - 1);
      #pragma unroll
      for (int i = 0; i < 8; ++i) {
        const int gn  = n0 + c + i;
        const int gnc = gn < Nn ? gn : (Nn - 1);
        const float v = BT ? Bm[(long long)gnc * ldb + gkc]
                           : Bm[(long long)gkc * ldb + gnc];
        Blds[r][c + i] = f2bf((gk < K && gn < Nn) ? v : 0.f);
      }
    }
    __syncthreads();
    {
      v8bf alo = *(const v8bf*)&Alds[mloc][half * 8];
      v8bf ahi = *(const v8bf*)&Alds[mloc][16 + half * 8];
      v16bf afrag = __builtin_shufflevector(alo, ahi,
          0,1,2,3,4,5,6,7,8,9,10,11,12,13,14,15);
      #pragma unroll
      for (int nt = 0; nt < 4; ++nt) {
        v16bf bfrag;
        #pragma unroll
        for (int s = 0; s < 16; ++s)
          bfrag[s] = Blds[s < 8 ? half * 8 + s : 16 + half * 8 + (s - 8)][nt * 16 + (lane & 15)];
        acc[nt] = __builtin_amdgcn_wmma_f32_16x16x32_bf16(
            false, afrag, false, bfrag, (short)0, acc[nt], false, false);
      }
    }
    __syncthreads();
  }

  // ---------------- epilogue + store ----------------
  const int row0 = m0 + wave * 16;
  if ((m0 + BM <= M) && nfull) {
    #pragma unroll
    for (int nt = 0; nt < 4; ++nt) {
      const int col = n0 + nt * 16 + (lane & 15);
      float* cp = Cm + (long long)(row0 + 8 * half) * ldc + col;
      #pragma unroll
      for (int r = 0; r < 8; ++r) {
        float v = acc[nt][r];
        if (EPI >= 2 && bias) v += bias[col];
        if (EPI == 2 && resid) v += resid[(long long)(row0 + r + 8 * half) * ldr + col];
        if (EPI == 3) v = 0.5f * v * (1.f + erff(v * 0.70710678118f));
        cp[(long long)r * ldc] = v;
      }
    }
  } else {
    #pragma unroll
    for (int nt = 0; nt < 4; ++nt) {
      const int col = n0 + nt * 16 + (lane & 15);
      if (col >= Nn) continue;
      #pragma unroll
      for (int r = 0; r < 8; ++r) {
        const int row = row0 + r + 8 * half;
        if (row >= M) continue;
        float v = acc[nt][r];
        if (EPI >= 2 && bias) v += bias[col];
        if (EPI == 2 && resid) v += resid[(long long)row * ldr + col];
        if (EPI == 3) v = 0.5f * v * (1.f + erff(v * 0.70710678118f));
        Cm[(long long)row * ldc + col] = v;
      }
    }
  }
}

// =====================================================================
// LayerNorm: one wave per row of length CDIM
// =====================================================================
__global__ __launch_bounds__(256)
void ln_kernel(const float* __restrict__ x, const float* __restrict__ w,
               const float* __restrict__ b, float* __restrict__ out, int rows)
{
  const int row  = blockIdx.x * 8 + (threadIdx.x >> 5);
  const int lane = threadIdx.x & 31;
  if (row >= rows) return;
  const float* xr = x + (long long)row * CDIM;
  float s = 0.f;
  for (int i = lane; i < CDIM; i += 32) s += xr[i];
  #pragma unroll
  for (int o = 16; o; o >>= 1) s += __shfl_xor(s, o, 32);
  const float mu = s * (1.f / CDIM);
  float var = 0.f;
  for (int i = lane; i < CDIM; i += 32) { float d = xr[i] - mu; var += d * d; }
  #pragma unroll
  for (int o = 16; o; o >>= 1) var += __shfl_xor(var, o, 32);
  const float rs = rsqrtf(var * (1.f / CDIM) + 1e-5f);
  float* orow = out + (long long)row * CDIM;
  for (int i = lane; i < CDIM; i += 32) orow[i] = (xr[i] - mu) * rs * w[i] + b[i];
}

// =====================================================================
// Split QKV [B*N,3C] -> q,k,v [B,H,N,hd]; q pre-scaled by hd^-0.5
// =====================================================================
__global__ void split_qkv(const float* __restrict__ qkv, float* __restrict__ q,
                          float* __restrict__ k, float* __restrict__ v,
                          float scale, long long total)
{
  long long i = (long long)blockIdx.x * 256 + threadIdx.x;
  if (i >= total) return;
  const int d = (int)(i % HD);
  long long t = i / HD;
  const int n = (int)(t % NTOK); t /= NTOK;
  const int h = (int)(t % NH);
  const int b = (int)(t / NH);
  const long long src = ((long long)(b * NTOK + n)) * (3 * CDIM) + h * HD + d;
  q[i] = qkv[src] * scale;
  k[i] = qkv[src + CDIM];
  v[i] = qkv[src + 2 * CDIM];
}

// =====================================================================
// Row softmax over attn rows of length NTOK (one wave per row)
// =====================================================================
__global__ __launch_bounds__(256)
void softmax_kernel(float* __restrict__ attn, long long rows)
{
  const long long row = (long long)blockIdx.x * 8 + (threadIdx.x >> 5);
  const int lane = threadIdx.x & 31;
  if (row >= rows) return;
  float* a = attn + row * NTOK;
  float mx = -1e30f;
  for (int i = lane; i < NTOK; i += 32) mx = fmaxf(mx, a[i]);
  #pragma unroll
  for (int o = 16; o; o >>= 1) mx = fmaxf(mx, __shfl_xor(mx, o, 32));
  float sum = 0.f;
  for (int i = lane; i < NTOK; i += 32) { float e = __expf(a[i] - mx); a[i] = e; sum += e; }
  #pragma unroll
  for (int o = 16; o; o >>= 1) sum += __shfl_xor(sum, o, 32);
  const float inv = 1.f / sum;
  for (int i = lane; i < NTOK; i += 32) a[i] *= inv;
}

// =====================================================================
// Histogram rank-selection: refine [lo, lo+width) so that roughly
// (rank+1) elements are >= lo.  Run twice -> ~1e-6 resolution on [0,1).
// =====================================================================
__global__ __launch_bounds__(256)
void select_kernel(const float* __restrict__ data, long long stride, int n,
                   int rank, float* __restrict__ lo_io, float* __restrict__ w_io)
{
  __shared__ int hist[2048];
  __shared__ float res[2];
  const int z = blockIdx.x;
  const float* d = data + (long long)z * stride;
  const float lo = lo_io[z], w = w_io[z];
  const float inv = 2048.f / w;
  for (int i = threadIdx.x; i < 2048; i += 256) hist[i] = 0;
  __syncthreads();
  for (int i = threadIdx.x; i < n; i += 256) {
    const int bin = (int)((d[i] - lo) * inv);
    if (bin >= 0) atomicAdd(&hist[bin > 2047 ? 2047 : bin], 1);
  }
  __syncthreads();
  if (threadIdx.x == 0) {
    int cum = 0, bsel = 0;
    for (int bi = 2047; bi >= 0; --bi) { cum += hist[bi]; if (cum > rank) { bsel = bi; break; } }
    res[0] = lo + bsel * (w / 2048.f);
    res[1] = w / 2048.f;
  }
  __syncthreads();
  if (threadIdx.x == 0) { lo_io[z] = res[0]; w_io[z] = res[1]; }
}

__global__ void init_sel(float* lo, float* w, int n) {
  const int i = blockIdx.x * 256 + threadIdx.x;
  if (i < n) { lo[i] = 0.f; w[i] = 1.f; }
}

// =====================================================================
// attn = (attn >= sigma) ? attn : 0   per (b,h)
// =====================================================================
__global__ void sparsify_attn(float* __restrict__ attn, const float* __restrict__ sigma)
{
  const int z = blockIdx.x;
  const float s = sigma[z];
  float* a = attn + (long long)z * NTOK * NTOK;
  for (int i = threadIdx.x; i < NTOK * NTOK; i += 256)
    if (a[i] < s) a[i] = 0.f;
}

// token_rank[b,t] = mean_h attn[b,h,t+1,t+1]
__global__ void token_rank_kernel(const float* __restrict__ attn, float* __restrict__ tr)
{
  const int b = blockIdx.x;
  for (int t = threadIdx.x; t < NTOK - 1; t += 256) {
    float s = 0.f;
    for (int h = 0; h < NH; ++h)
      s += attn[(((long long)(b * NH + h)) * NTOK + (t + 1)) * NTOK + (t + 1)];
    tr[b * (NTOK - 1) + t] = s * (1.f / NH);
  }
}

// =====================================================================
// Per-batch descending bitonic argsort of 196 values (padded to 256)
// =====================================================================
__global__ __launch_bounds__(256)
void argsort_kernel(const float* __restrict__ tr, int* __restrict__ ik, int* __restrict__ ie)
{
  __shared__ float val[256];
  __shared__ int   ind[256];
  const int b = blockIdx.x, t = threadIdx.x;
  val[t] = (t < NTOK - 1) ? tr[b * (NTOK - 1) + t] : -1e30f;
  ind[t] = t;
  __syncthreads();
  for (int ks = 2; ks <= 256; ks <<= 1) {
    for (int j = ks >> 1; j > 0; j >>= 1) {
      const int ixj = t ^ j;
      if (ixj > t) {
        const bool descending = ((t & ks) == 0);
        const bool doswap = descending ? (val[t] < val[ixj]) : (val[t] > val[ixj]);
        if (doswap) {
          float fv = val[t]; val[t] = val[ixj]; val[ixj] = fv;
          int   ii = ind[t]; ind[t] = ind[ixj]; ind[ixj] = ii;
        }
      }
      __syncthreads();
    }
  }
  if (t < NKEPT)               ik[b * NKEPT + t] = ind[t] + 1;
  else if (t < NTOK - 1)       ie[b * NPROP + (t - NKEPT)] = ind[t] + 1;
}

// av [B,H,N,hd] -> attnout [B,N,C]
__global__ void merge_heads(const float* __restrict__ av, float* __restrict__ out, long long total)
{
  long long i = (long long)blockIdx.x * 256 + threadIdx.x;
  if (i >= total) return;
  const int d = (int)(i % HD);
  long long t = i / HD;
  const int n = (int)(t % NTOK); t /= NTOK;
  const int h = (int)(t % NH);
  const int b = (int)(t / NH);
  out[((long long)(b * NTOK + n)) * CDIM + h * HD + d] = av[i];
}

// w[b,h,k,e] = attn[b,h, kept[k], elim[e]]
__global__ void gather_w(const float* __restrict__ attn, const int* __restrict__ ik,
                         const int* __restrict__ ie, float* __restrict__ wbuf)
{
  const int z = blockIdx.x;      // b*NH + h
  const int b = z / NH;
  const float* a = attn + (long long)z * NTOK * NTOK;
  float* wz = wbuf + (long long)z * NKEPT * NPROP;
  for (int i = threadIdx.x; i < NKEPT * NPROP; i += 256) {
    const int kk = i / NPROP, e = i % NPROP;
    wz[i] = a[(long long)ik[b * NKEPT + kk] * NTOK + ie[b * NPROP + e]];
  }
}

// x_prop[b,k,h*64+d] = sum_e thr(w[b,h,k,e]) * xattn[b, elim[e], h*64+d]
__global__ __launch_bounds__(256)
void xprop_kernel(const float* __restrict__ wbuf, const float* __restrict__ wsig,
                  const float* __restrict__ xattn, const int* __restrict__ ie,
                  float* __restrict__ xprop)
{
  __shared__ float wl[NKEPT * NPROP];   // 24.4 KB
  __shared__ float xe[NPROP][HD];       // 10 KB
  const int z = blockIdx.x;
  const int b = z / NH, h = z % NH;
  const float sig = wsig[z];
  const float* wz = wbuf + (long long)z * NKEPT * NPROP;
  for (int i = threadIdx.x; i < NKEPT * NPROP; i += 256) {
    const float v = wz[i];
    wl[i] = (v >= sig) ? v : 0.f;
  }
  for (int i = threadIdx.x; i < NPROP * HD; i += 256) {
    const int e = i / HD, d = i % HD;
    xe[e][d] = xattn[((long long)(b * NTOK + ie[b * NPROP + e])) * CDIM + h * HD + d];
  }
  __syncthreads();
  for (int i = threadIdx.x; i < NKEPT * HD; i += 256) {
    const int kk = i / HD, d = i % HD;
    float acc = 0.f;
    #pragma unroll 8
    for (int e = 0; e < NPROP; ++e) acc += wl[kk * NPROP + e] * xe[e][d];
    xprop[((long long)(b * NKEPT + kk)) * CDIM + h * HD + d] = acc;
  }
}

// x2[b,0,:]=xattn[b,0,:]; x2[b,1+k,:]=xattn[b,kept[k],:]+ALPHA*xprop[b,k,:]
__global__ void build_x2(const float* __restrict__ xattn, const float* __restrict__ xprop,
                         const int* __restrict__ ik, float* __restrict__ x2, long long total)
{
  long long i = (long long)blockIdx.x * 256 + threadIdx.x;
  if (i >= total) return;
  const int c = (int)(i % CDIM);
  long long t = i / CDIM;
  const int tt = (int)(t % NPRUNE);
  const int b  = (int)(t / NPRUNE);
  if (tt == 0) {
    x2[i] = xattn[((long long)(b * NTOK)) * CDIM + c];
  } else {
    const int kk = tt - 1;
    x2[i] = xattn[((long long)(b * NTOK + ik[b * NKEPT + kk])) * CDIM + c]
          + ALPHA * xprop[((long long)(b * NKEPT + kk)) * CDIM + c];
  }
}

// =====================================================================
// Orchestration
// =====================================================================
extern "C" void kernel_launch(void* const* d_in, const int* in_sizes, int n_in,
                              void* d_out, int out_size, void* d_ws, size_t ws_size,
                              hipStream_t stream)
{
  const float* x     = (const float*)d_in[0];
  const float* n1w   = (const float*)d_in[1];
  const float* n1b   = (const float*)d_in[2];
  const float* qkvw  = (const float*)d_in[3];
  const float* projw = (const float*)d_in[4];
  const float* projb = (const float*)d_in[5];
  const float* n2w   = (const float*)d_in[6];
  const float* n2b   = (const float*)d_in[7];
  const float* fc1w  = (const float*)d_in[8];
  const float* fc1b  = (const float*)d_in[9];
  const float* fc2w  = (const float*)d_in[10];
  const float* fc2b  = (const float*)d_in[11];
  float* out = (float*)d_out;

  const int ROWS1 = BSZ * NTOK;     // 12608
  const int ROWS2 = BSZ * NPRUNE;   // 10048
  const int BH    = BSZ * NH;       // 768

  // bump allocator over workspace
  char* wp = (char*)d_ws;
  auto alloc = [&](size_t bytes) -> void* {
    void* p = (void*)wp;
    wp += (bytes + 255) & ~(size_t)255;
    return p;
  };
  float* hln1    = (float*)alloc((size_t)ROWS1 * CDIM * 4);
  float* qkv     = (float*)alloc((size_t)ROWS1 * 3 * CDIM * 4);
  float* qb      = (float*)alloc((size_t)BH * NTOK * HD * 4);
  float* kb      = (float*)alloc((size_t)BH * NTOK * HD * 4);
  float* vb      = (float*)alloc((size_t)BH * NTOK * HD * 4);
  float* attn    = (float*)alloc((size_t)BH * NTOK * NTOK * 4);
  float* sel_lo  = (float*)alloc((size_t)BH * 4);
  float* sel_w   = (float*)alloc((size_t)BH * 4);
  float* trank   = (float*)alloc((size_t)BSZ * (NTOK - 1) * 4);
  int*   ik      = (int*)  alloc((size_t)BSZ * NKEPT * 4);
  int*   ie      = (int*)  alloc((size_t)BSZ * NPROP * 4);
  float* av      = (float*)alloc((size_t)BH * NTOK * HD * 4);
  float* attnout = (float*)alloc((size_t)ROWS1 * CDIM * 4);
  float* xattn   = (float*)alloc((size_t)ROWS1 * CDIM * 4);
  float* wbuf    = (float*)alloc((size_t)BH * NKEPT * NPROP * 4);
  float* xprop   = (float*)alloc((size_t)BSZ * NKEPT * CDIM * 4);
  float* x2      = (float*)alloc((size_t)ROWS2 * CDIM * 4);
  float* hln2    = (float*)alloc((size_t)ROWS2 * CDIM * 4);
  float* fc1o    = (float*)alloc((size_t)ROWS2 * MLPH * 4);

  auto cdiv = [](long long a, long long b) { return (unsigned)((a + b - 1) / b); };

  // 1) LN1
  ln_kernel<<<cdiv(ROWS1, 8), 256, 0, stream>>>(x, n1w, n1b, hln1, ROWS1);

  // 2) QKV GEMM: [12608,768] x [2304,768]^T
  gemm_wmma<0, true, true><<<dim3(cdiv(3 * CDIM, BN), cdiv(ROWS1, BM), 1), 256, 0, stream>>>(
      hln1, 0, CDIM, qkvw, 0, CDIM, qkv, 0, 3 * CDIM,
      nullptr, nullptr, 0, ROWS1, 3 * CDIM, CDIM);

  // 3) split q,k,v (+scale q)
  {
    long long total = (long long)BH * NTOK * HD;
    split_qkv<<<cdiv(total, 256), 256, 0, stream>>>(qkv, qb, kb, vb, 0.125f, total);
  }

  // 4) S = q @ k^T  (batched 768x: 197x197, K=64)
  gemm_wmma<0, true, true><<<dim3(cdiv(NTOK, BN), cdiv(NTOK, BM), BH), 256, 0, stream>>>(
      qb, (long long)NTOK * HD, HD, kb, (long long)NTOK * HD, HD,
      attn, (long long)NTOK * NTOK, NTOK,
      nullptr, nullptr, 0, NTOK, NTOK, HD);

  // 5) softmax rows
  softmax_kernel<<<cdiv((long long)BH * NTOK, 8), 256, 0, stream>>>(attn, (long long)BH * NTOK);

  // 6) per-(b,h) 75%-keep threshold via 2-level histogram selection
  {
    const int rank = (int)((long long)NTOK * NTOK * 3 / 4);   // int(N*N*0.75) = 29106
    init_sel<<<cdiv(BH, 256), 256, 0, stream>>>(sel_lo, sel_w, BH);
    select_kernel<<<BH, 256, 0, stream>>>(attn, (long long)NTOK * NTOK, NTOK * NTOK, rank, sel_lo, sel_w);
    select_kernel<<<BH, 256, 0, stream>>>(attn, (long long)NTOK * NTOK, NTOK * NTOK, rank, sel_lo, sel_w);
    sparsify_attn<<<BH, 256, 0, stream>>>(attn, sel_lo);
  }

  // 7) token ranking + argsort
  token_rank_kernel<<<BSZ, 256, 0, stream>>>(attn, trank);
  argsort_kernel<<<BSZ, 256, 0, stream>>>(trank, ik, ie);

  // 8) av = attn @ v  (batched 768x: 197x64, K=197; lda=197 not vec4)
  gemm_wmma<0, false, false><<<dim3(cdiv(HD, BN), cdiv(NTOK, BM), BH), 256, 0, stream>>>(
      attn, (long long)NTOK * NTOK, NTOK, vb, (long long)NTOK * HD, HD,
      av, (long long)NTOK * HD, HD,
      nullptr, nullptr, 0, NTOK, HD, NTOK);

  // 9) merge heads
  {
    long long total = (long long)BH * NTOK * HD;
    merge_heads<<<cdiv(total, 256), 256, 0, stream>>>(av, attnout, total);
  }

  // 10) xattn = x + attnout @ proj_w^T + proj_b
  gemm_wmma<2, true, true><<<dim3(cdiv(CDIM, BN), cdiv(ROWS1, BM), 1), 256, 0, stream>>>(
      attnout, 0, CDIM, projw, 0, CDIM, xattn, 0, CDIM,
      projb, x, CDIM, ROWS1, CDIM, CDIM);

  // 11) gather w, sparsify (top 5%), propagate
  gather_w<<<BH, 256, 0, stream>>>(attn, ik, ie, wbuf);
  {
    const int rank = (NKEPT * NPROP) / 20;                    // int(6240*0.05) = 312
    init_sel<<<cdiv(BH, 256), 256, 0, stream>>>(sel_lo, sel_w, BH);
    select_kernel<<<BH, 256, 0, stream>>>(wbuf, (long long)NKEPT * NPROP, NKEPT * NPROP, rank, sel_lo, sel_w);
    select_kernel<<<BH, 256, 0, stream>>>(wbuf, (long long)NKEPT * NPROP, NKEPT * NPROP, rank, sel_lo, sel_w);
  }
  xprop_kernel<<<BH, 256, 0, stream>>>(wbuf, sel_lo, xattn, ie, xprop);

  // 12) assemble pruned sequence
  {
    long long total = (long long)ROWS2 * CDIM;
    build_x2<<<cdiv(total, 256), 256, 0, stream>>>(xattn, xprop, ik, x2, total);
  }

  // 13) LN2 + MLP
  ln_kernel<<<cdiv(ROWS2, 8), 256, 0, stream>>>(x2, n2w, n2b, hln2, ROWS2);

  gemm_wmma<3, true, true><<<dim3(cdiv(MLPH, BN), cdiv(ROWS2, BM), 1), 256, 0, stream>>>(
      hln2, 0, CDIM, fc1w, 0, CDIM, fc1o, 0, MLPH,
      fc1b, nullptr, 0, ROWS2, MLPH, CDIM);

  gemm_wmma<2, true, true><<<dim3(cdiv(CDIM, BN), cdiv(ROWS2, BM), 1), 256, 0, stream>>>(
      fc1o, 0, MLPH, fc2w, 0, MLPH, out, 0, CDIM,
      fc2b, x2, CDIM, ROWS2, CDIM, MLPH);
}
